// IgformerModel_13108240188119
// MI455X (gfx1250) — matrix operand order repaired and checked by reading, not compiled, not run on hardware
//
#include <hip/hip_runtime.h>
#include <hip/hip_bf16.h>

// ---------------- problem constants ----------------
#define NNODES 50000
#define CCH 256
#define NH 4
#define DH 256
#define HDIM 1024
#define NLAYERS 2
#define LN_EPS 1e-5f

// ---------------- WMMA tiling ----------------
typedef __attribute__((ext_vector_type(16))) __bf16 v16bf;
typedef __attribute__((ext_vector_type(8)))  float  v8f;

#define BM 128
#define BN 128
#define BK 64    // two 16x16x32 K-steps per staged tile
#define LDA 72   // LDS leading dim (halves); 144B rows: 16B-aligned, bank-friendly
#define LDB 72
#define KCHUNK 2048
#define RPB 512

__device__ __forceinline__ v8f wmma_bf16(v16bf a, v16bf b, v8f c) {
  return __builtin_amdgcn_wmma_f32_16x16x32_bf16(false, a, false, b, (short)0, c,
                                                 false, false);
}

// A fragment (16x32) from LDS tile As[m][k] (ld=LDA), K window at koff.
// lanes 0-15: M=lane, K{0..7,16..23}; lanes 16-31: M=lane-16, K{8..15,24..31}.
// Both 8-half runs contiguous -> 2x ds_load_b128 per lane.
__device__ __forceinline__ v16bf load_A_frag(const __bf16* As, int mBase,
                                             int koff, int lane) {
  const __bf16* p =
      As + (mBase + (lane & 15)) * LDA + koff + ((lane < 16) ? 0 : 8);
  v16bf a;
#pragma unroll
  for (int j = 0; j < 8; ++j) a[j] = p[j];
#pragma unroll
  for (int j = 8; j < 16; ++j) a[j] = p[j + 8];
  return a;
}

// B fragment (32x16) from LDS tile Bs[n][k] (ld=LDB), K window at koff.
// lanes 0-15: N=lane, K{0..7,16..23}; lanes 16-31: N=lane-16, K{8..15,24..31}.
// Both 8-half runs contiguous -> 2x ds_load_b128 per lane.
__device__ __forceinline__ v16bf load_B_frag(const __bf16* Bs, int nBase,
                                             int koff, int lane) {
  const __bf16* p =
      Bs + (nBase + (lane & 15)) * LDB + koff + ((lane < 16) ? 0 : 8);
  v16bf b;
#pragma unroll
  for (int j = 0; j < 8; ++j) b[j] = p[j];
#pragma unroll
  for (int j = 8; j < 16; ++j) b[j] = p[j + 8];
  return b;
}

__device__ __forceinline__ void copy8(__bf16* dst, const __bf16* src) {
  *(ulonglong2*)dst = *(const ulonglong2*)src;  // 16-byte move
}
__device__ __forceinline__ void zero8(__bf16* dst) {
  ulonglong2 z; z.x = 0ull; z.y = 0ull;
  *(ulonglong2*)dst = z;
}
union U8 { ulonglong2 v; __bf16 h[8]; };
__device__ __forceinline__ void store_t8(__bf16* S, int n0, int k, int ld,
                                         const __bf16* src) {
  U8 u; u.v = *(const ulonglong2*)src;
#pragma unroll
  for (int i = 0; i < 8; ++i) S[(n0 + i) * ld + k] = u.h[i];
}

// ---------------- tile staging (128x64 halves, 256 threads, 4 chunks ea) ----------------
// A row-major [M,K]: k-contiguous -> vector stores into As[m][k]
__device__ __forceinline__ void stage_A_rows(__bf16* As, const __bf16* A,
                                             int rowBase, int kt, int M, int K,
                                             int tid) {
#pragma unroll
  for (int u = 0; u < 4; ++u) {
    int e8 = tid + u * 256;
    int r = e8 >> 3;          // 8 chunks per 64-half row
    int c = (e8 & 7) * 8;
    __bf16* dst = As + r * LDA + c;
    if (rowBase + r < M) {
      const __bf16* src = A + (size_t)(rowBase + r) * K + kt + c;
      copy8(dst, src);
      if (kt + 2 * BK < K)
        __builtin_prefetch((const void*)(src + 2 * BK), 0, 0);
    } else {
      zero8(dst);
    }
  }
}
// BT row-major [N,K] (pre-transposed weights): k-contiguous -> vector stores
__device__ __forceinline__ void stage_B_rows(__bf16* Bs, const __bf16* BT,
                                             int colBase, int kt, int K,
                                             int tid) {
#pragma unroll
  for (int u = 0; u < 4; ++u) {
    int e8 = tid + u * 256;
    int n = e8 >> 3;
    int c = (e8 & 7) * 8;
    copy8(Bs + n * LDB + c, BT + (size_t)(colBase + n) * K + kt + c);
  }
}
// node-major [Nrows, HDIM] source, transposed into S[mn][k] (k = node index)
__device__ __forceinline__ void stage_T(__bf16* S, int ld, const __bf16* mat,
                                        int base, int h, int kt, int Nrows,
                                        int tid) {
#pragma unroll
  for (int u = 0; u < 4; ++u) {
    int e8 = tid + u * 256;
    int r = e8 >> 4;          // k row within tile (node), 0..63
    int c = (e8 & 15) * 8;    // m/n chunk
    int gr = kt + r;
    if (gr < Nrows) {
      store_t8(S, c, r, ld, mat + (size_t)gr * HDIM + h * DH + base + c);
    } else {
#pragma unroll
      for (int i = 0; i < 8; ++i) S[(c + i) * ld + r] = (__bf16)0.f;
    }
  }
}

// All four B fragments live simultaneously so their ds_load_b128s issue
// back-to-back instead of the reuse-register wait-serialized pattern.
__device__ __forceinline__ void mma_tile(const __bf16* As, const __bf16* Bs,
                                         v8f (&acc)[2][4], int wm, int wn,
                                         int lane) {
#pragma unroll
  for (int kk = 0; kk < BK; kk += 32) {
    v16bf a0 = load_A_frag(As, wm * 32, kk, lane);
    v16bf a1 = load_A_frag(As, wm * 32 + 16, kk, lane);
    v16bf b0 = load_B_frag(Bs, wn * 64, kk, lane);
    v16bf b1 = load_B_frag(Bs, wn * 64 + 16, kk, lane);
    v16bf b2 = load_B_frag(Bs, wn * 64 + 32, kk, lane);
    v16bf b3 = load_B_frag(Bs, wn * 64 + 48, kk, lane);
    acc[0][0] = wmma_bf16(a0, b0, acc[0][0]);
    acc[1][0] = wmma_bf16(a1, b0, acc[1][0]);
    acc[0][1] = wmma_bf16(a0, b1, acc[0][1]);
    acc[1][1] = wmma_bf16(a1, b1, acc[1][1]);
    acc[0][2] = wmma_bf16(a0, b2, acc[0][2]);
    acc[1][2] = wmma_bf16(a1, b2, acc[1][2]);
    acc[0][3] = wmma_bf16(a0, b3, acc[0][3]);
    acc[1][3] = wmma_bf16(a1, b3, acc[1][3]);
  }
}

// ------------- GEMM: out[M,N] = A[M,K] @ W[K,N] + bias (W given transposed) -------------
__global__ __launch_bounds__(256) void k_gemm_bf16(
    const __bf16* __restrict__ A, const __bf16* __restrict__ BT,
    const float* __restrict__ bias, float* __restrict__ outF,
    __bf16* __restrict__ outB, int M, int N, int K) {
  __shared__ __attribute__((aligned(16))) __bf16 As[2][BM * LDA];
  __shared__ __attribute__((aligned(16))) __bf16 Bs[2][BN * LDB];
  const int tid = threadIdx.x;
  const int lane = tid & 31;
  const int w = tid >> 5;
  const int wm = w >> 1;
  const int wn = w & 1;
  const int rowBase = blockIdx.x * BM;
  const int colBase = blockIdx.y * BN;

  v8f acc[2][4] = {};
  stage_A_rows(As[0], A, rowBase, 0, M, K, tid);
  stage_B_rows(Bs[0], BT, colBase, 0, K, tid);
  int cur = 0;
  const int nt = K / BK;
  for (int t = 0; t < nt; ++t) {
    __syncthreads();
    if (t + 1 < nt) {
      stage_A_rows(As[cur ^ 1], A, rowBase, (t + 1) * BK, M, K, tid);
      stage_B_rows(Bs[cur ^ 1], BT, colBase, (t + 1) * BK, K, tid);
    }
    mma_tile(As[cur], Bs[cur], acc, wm, wn, lane);
    cur ^= 1;
  }

  const int lm = (lane < 16) ? 0 : 8;
  const int ln_ = lane & 15;
#pragma unroll
  for (int i = 0; i < 2; ++i)
#pragma unroll
    for (int j = 0; j < 4; ++j) {
      int col = colBase + wn * 64 + j * 16 + ln_;
      float bv = bias ? bias[col] : 0.f;
#pragma unroll
      for (int r = 0; r < 8; ++r) {
        int row = rowBase + wm * 32 + i * 16 + lm + r;
        if (row < M) {
          float v = acc[i][j][r] + bv;
          if (outF) outF[(size_t)row * N + col] = v;
          if (outB) outB[(size_t)row * N + col] = (__bf16)v;
        }
      }
    }
}

// ------------- kvsT[h][n][m] += k_h^T @ v_h  (split-K, f32 atomics) -------------
__global__ __launch_bounds__(256) void k_kvs(
    const __bf16* __restrict__ kmat, const __bf16* __restrict__ vmat,
    float* __restrict__ kvsFT, int Nrows) {
  __shared__ __attribute__((aligned(16))) __bf16 As[2][BM * LDA];
  __shared__ __attribute__((aligned(16))) __bf16 Bs[2][BN * LDB];
  const int tid = threadIdx.x;
  const int lane = tid & 31;
  const int w = tid >> 5;
  const int wm = w >> 1;
  const int wn = w & 1;
  const int mBase = (blockIdx.x >> 1) * BM;
  const int nBase = (blockIdx.x & 1) * BN;
  const int h = blockIdx.y;
  const int k0 = blockIdx.z * KCHUNK;

  v8f acc[2][4] = {};
  stage_T(As[0], LDA, kmat, mBase, h, k0, Nrows, tid);
  stage_T(Bs[0], LDB, vmat, nBase, h, k0, Nrows, tid);
  int cur = 0;
  const int nt = KCHUNK / BK;
  for (int t = 0; t < nt; ++t) {
    __syncthreads();
    if (t + 1 < nt) {
      int kt = k0 + (t + 1) * BK;
      stage_T(As[cur ^ 1], LDA, kmat, mBase, h, kt, Nrows, tid);
      stage_T(Bs[cur ^ 1], LDB, vmat, nBase, h, kt, Nrows, tid);
    }
    mma_tile(As[cur], Bs[cur], acc, wm, wn, lane);
    cur ^= 1;
  }

  const int lm = (lane < 16) ? 0 : 8;
  const int ln_ = lane & 15;
#pragma unroll
  for (int i = 0; i < 2; ++i)
#pragma unroll
    for (int j = 0; j < 4; ++j) {
      int col = nBase + wn * 64 + j * 16 + ln_;
#pragma unroll
      for (int r = 0; r < 8; ++r) {
        int row = mBase + wm * 32 + i * 16 + lm + r;
        atomicAdd(&kvsFT[(size_t)h * DH * DH + (size_t)col * DH + row],
                  acc[i][j][r]);
      }
    }
}

// ------ attn: out = alpha*mean_h[(s*q@kvs_h + N*v_h)/denom_h] + (1-alpha)*prev ------
__device__ __forceinline__ void stage_Q(__bf16* As, const __bf16* q, int rowBase,
                                        int h, int kt, int Nrows, int tid) {
#pragma unroll
  for (int u = 0; u < 4; ++u) {
    int e8 = tid + u * 256;
    int r = e8 >> 3;
    int c = (e8 & 7) * 8;
    __bf16* dst = As + r * LDA + c;
    if (rowBase + r < Nrows)
      copy8(dst, q + (size_t)(rowBase + r) * HDIM + h * DH + kt + c);
    else
      zero8(dst);
  }
}
__device__ __forceinline__ void stage_KVS(__bf16* Bs, const __bf16* kvsBT,
                                          int colBase, int h, int kt, int tid) {
#pragma unroll
  for (int u = 0; u < 4; ++u) {
    int e8 = tid + u * 256;
    int n = e8 >> 3;
    int c = (e8 & 7) * 8;
    copy8(Bs + n * LDB + c,
          kvsBT + (size_t)h * DH * DH + (size_t)(colBase + n) * DH + kt + c);
  }
}

__global__ __launch_bounds__(256) void k_attn(
    const __bf16* __restrict__ q, const __bf16* __restrict__ v,
    const __bf16* __restrict__ kvsBT, const float* __restrict__ denomBuf,
    const float* __restrict__ ss, const float* __restrict__ prevF,
    float* __restrict__ attnF, int Nrows) {
  __shared__ __attribute__((aligned(16))) __bf16 As[2][BM * LDA];
  __shared__ __attribute__((aligned(16))) __bf16 Bs[2][BN * LDB];
  const int tid = threadIdx.x;
  const int lane = tid & 31;
  const int w = tid >> 5;
  const int wm = w >> 1;
  const int wn = w & 1;
  const int rowBase = blockIdx.x * BM;
  const int colBase = blockIdx.y * BN;
  const float s = rsqrtf(ss[0]) * rsqrtf(ss[1]);  // 1/(||q||*||k||)
  const int lm = (lane < 16) ? 0 : 8;
  const int ln_ = lane & 15;

  float outAcc[2][4][8] = {};
  v8f acc[2][4] = {};
  stage_Q(As[0], q, rowBase, 0, 0, Nrows, tid);
  stage_KVS(Bs[0], kvsBT, colBase, 0, 0, tid);
  int cur = 0;
  const int KT_PER_H = DH / BK;                   // 4
  const int NT = NH * KT_PER_H;                   // 16 pipelined tiles
  for (int t = 0; t < NT; ++t) {
    __syncthreads();
    if (t + 1 < NT) {
      int h2 = (t + 1) >> 2;
      int kt2 = ((t + 1) & 3) * BK;
      stage_Q(As[cur ^ 1], q, rowBase, h2, kt2, Nrows, tid);
      stage_KVS(Bs[cur ^ 1], kvsBT, colBase, h2, kt2, tid);
    }
    mma_tile(As[cur], Bs[cur], acc, wm, wn, lane);
    if ((t & 3) == 3) {  // end of head: fold into output accumulator
      int h = t >> 2;
#pragma unroll
      for (int i = 0; i < 2; ++i)
#pragma unroll
        for (int j = 0; j < 4; ++j) {
          int col = colBase + wn * 64 + j * 16 + ln_;
#pragma unroll
          for (int r = 0; r < 8; ++r) {
            int row = rowBase + wm * 32 + i * 16 + lm + r;
            if (row < Nrows) {
              float dn = denomBuf[(size_t)row * NH + h];
              float vv = (float)v[(size_t)row * HDIM + h * DH + col];
              outAcc[i][j][r] += (s * acc[i][j][r] + (float)Nrows * vv) / dn;
            }
            acc[i][j][r] = 0.f;
          }
        }
    }
    cur ^= 1;
  }

#pragma unroll
  for (int i = 0; i < 2; ++i)
#pragma unroll
    for (int j = 0; j < 4; ++j) {
      int col = colBase + wn * 64 + j * 16 + ln_;
#pragma unroll
      for (int r = 0; r < 8; ++r) {
        int row = rowBase + wm * 32 + i * 16 + lm + r;
        if (row < Nrows) {
          float attn = outAcc[i][j][r] * (1.0f / NH);
          attnF[(size_t)row * CCH + col] =
              0.5f * attn + 0.5f * prevF[(size_t)row * CCH + col];
        }
      }
    }
}

// ---------------- small kernels ----------------
__global__ void k_f32_to_bf16(const float* __restrict__ src,
                              __bf16* __restrict__ dst, int n) {
  int i = blockIdx.x * blockDim.x + threadIdx.x;
  if (i < n) dst[i] = (__bf16)src[i];
}
// transpose-convert: src [K,N] f32 -> dst [N,K] bf16
__global__ void k_f32_to_bf16_T(const float* __restrict__ src,
                                __bf16* __restrict__ dst, int K, int N) {
  int i = blockIdx.x * blockDim.x + threadIdx.x;
  if (i < N * K) {
    int n = i / K;
    int k = i - n * K;
    dst[i] = (__bf16)src[(size_t)k * N + n];
  }
}
__global__ void k_zero_f32(float* __restrict__ p, int n) {
  int i = blockIdx.x * blockDim.x + threadIdx.x;
  if (i < n) p[i] = 0.f;
}

__global__ __launch_bounds__(256) void k_sumsq(const __bf16* __restrict__ p,
                                               long n, float* __restrict__ out) {
  __shared__ float red[256];
  float acc = 0.f;
  for (long i = (long)blockIdx.x * blockDim.x + threadIdx.x; i < n;
       i += (long)gridDim.x * blockDim.x) {
    float v = (float)p[i];
    acc += v * v;
  }
  red[threadIdx.x] = acc;
  __syncthreads();
  for (int st = 128; st > 0; st >>= 1) {
    if ((int)threadIdx.x < st) red[threadIdx.x] += red[threadIdx.x + st];
    __syncthreads();
  }
  if (threadIdx.x == 0) atomicAdd(out, red[0]);
}

__global__ __launch_bounds__(256) void k_colsum(const __bf16* __restrict__ kmat,
                                                int rows,
                                                float* __restrict__ ksum) {
  int r0 = blockIdx.x * RPB;
  int r1 = r0 + RPB; if (r1 > rows) r1 = rows;
  int c = threadIdx.x;
  float a0 = 0.f, a1 = 0.f, a2 = 0.f, a3 = 0.f;
  for (int r = r0; r < r1; ++r) {
    const __bf16* row = kmat + (size_t)r * HDIM;
    a0 += (float)row[c];
    a1 += (float)row[c + 256];
    a2 += (float)row[c + 512];
    a3 += (float)row[c + 768];
  }
  atomicAdd(&ksum[c],       a0);
  atomicAdd(&ksum[c + 256], a1);
  atomicAdd(&ksum[c + 512], a2);
  atomicAdd(&ksum[c + 768], a3);
}

__global__ __launch_bounds__(256) void k_denom(
    const __bf16* __restrict__ q, const float* __restrict__ ksum,
    const float* __restrict__ ss, float* __restrict__ denom, int nrows) {
  int t = blockIdx.x * 256 + threadIdx.x;
  int node = t >> 2, h = t & 3;
  if (node >= nrows) return;
  float s = rsqrtf(ss[0]) * rsqrtf(ss[1]);
  const __bf16* qp = q + (size_t)node * HDIM + h * DH;
  const float* kp = ksum + h * DH;
  float acc = 0.f;
  for (int d = 0; d < DH; ++d) acc += (float)qp[d] * kp[d];
  denom[(size_t)node * NH + h] = s * acc + (float)nrows;
}

__global__ __launch_bounds__(256) void k_layernorm(
    const float* __restrict__ in, const float* __restrict__ gw,
    const float* __restrict__ bw, float* __restrict__ outF,
    __bf16* __restrict__ outB, int relu) {
  __shared__ float red[256];
  int node = blockIdx.x;
  int c = threadIdx.x;
  float v = in[(size_t)node * CCH + c];
  red[c] = v;
  __syncthreads();
  for (int st = 128; st > 0; st >>= 1) {
    if (c < st) red[c] += red[c + st];
    __syncthreads();
  }
  float mu = red[0] * (1.0f / CCH);
  __syncthreads();
  float d = v - mu;
  red[c] = d * d;
  __syncthreads();
  for (int st = 128; st > 0; st >>= 1) {
    if (c < st) red[c] += red[c + st];
    __syncthreads();
  }
  float var = red[0] * (1.0f / CCH);
  float y = d * rsqrtf(var + LN_EPS) * gw[c] + bw[c];
  if (relu) y = fmaxf(y, 0.f);
  outF[(size_t)node * CCH + c] = y;
  outB[(size_t)node * CCH + c] = (__bf16)y;
}

// ---------------- driver ----------------
extern "C" void kernel_launch(void* const* d_in, const int* in_sizes, int n_in,
                              void* d_out, int out_size, void* d_ws,
                              size_t ws_size, hipStream_t stream) {
  (void)in_sizes; (void)n_in; (void)out_size; (void)ws_size;
  const float* x     = (const float*)d_in[0];
  const float* fc0_w = (const float*)d_in[1];
  const float* fc0_b = (const float*)d_in[2];
  const float* ln_w  = (const float*)d_in[3];
  const float* ln_b  = (const float*)d_in[4];
  const float* Wq    = (const float*)d_in[5];
  const float* bq    = (const float*)d_in[6];
  const float* Wk    = (const float*)d_in[7];
  const float* bk    = (const float*)d_in[8];
  const float* Wv    = (const float*)d_in[9];
  const float* bv    = (const float*)d_in[10];
  const float* fc_w  = (const float*)d_in[11];
  const float* fc_b  = (const float*)d_in[12];
  float* out = (float*)d_out;

  char* p = (char*)d_ws;
  auto alloc = [&](size_t bytes) -> void* {
    void* r = (void*)p;
    p += (bytes + 255) & ~(size_t)255;
    return r;
  };
  float*  hF    = (float*) alloc((size_t)NNODES * CCH * 4);
  float*  attnF = (float*) alloc((size_t)NNODES * CCH * 4);
  __bf16* hB    = (__bf16*)alloc((size_t)NNODES * CCH * 2);
  __bf16* qB    = (__bf16*)alloc((size_t)NNODES * HDIM * 2);
  __bf16* kB    = (__bf16*)alloc((size_t)NNODES * HDIM * 2);
  __bf16* vB    = (__bf16*)alloc((size_t)NNODES * HDIM * 2);
  float*  denom = (float*) alloc((size_t)NNODES * NH * 4);
  float*  kvsFT = (float*) alloc((size_t)NH * DH * DH * 4);
  __bf16* kvsBT = (__bf16*)alloc((size_t)NH * DH * DH * 2);
  float*  ksum  = (float*) alloc((size_t)HDIM * 4);
  float*  ss    = (float*) alloc(2 * 4);
  __bf16* fc0BT = (__bf16*)alloc((size_t)CCH * CCH * 2);   // [N][K]
  __bf16* fcBT  = (__bf16*)alloc((size_t)CCH * CCH * 2);   // [N][K]
  __bf16* WqBT  = (__bf16*)alloc((size_t)NLAYERS * CCH * HDIM * 2);  // [HD][C]
  __bf16* WkBT  = (__bf16*)alloc((size_t)NLAYERS * CCH * HDIM * 2);
  __bf16* WvBT  = (__bf16*)alloc((size_t)NLAYERS * CCH * HDIM * 2);

  dim3 blk(256);
  auto cvt = [&](const float* src, __bf16* dst, int n) {
    k_f32_to_bf16<<<dim3((n + 255) / 256), blk, 0, stream>>>(src, dst, n);
  };
  auto cvtT = [&](const float* src, __bf16* dst, int K, int N) {
    k_f32_to_bf16_T<<<dim3((N * K + 255) / 256), blk, 0, stream>>>(src, dst, K, N);
  };
  cvt(x, hB, NNODES * CCH);
  cvtT(fc0_w, fc0BT, CCH, CCH);
  cvtT(fc_w, fcBT, CCH, CCH);
  for (int l = 0; l < NLAYERS; ++l) {
    cvtT(Wq + (size_t)l * CCH * HDIM, WqBT + (size_t)l * HDIM * CCH, CCH, HDIM);
    cvtT(Wk + (size_t)l * CCH * HDIM, WkBT + (size_t)l * HDIM * CCH, CCH, HDIM);
    cvtT(Wv + (size_t)l * CCH * HDIM, WvBT + (size_t)l * HDIM * CCH, CCH, HDIM);
  }

  dim3 gSmall((NNODES + BM - 1) / BM, CCH / BN);   // (391, 2)
  dim3 gProj((NNODES + BM - 1) / BM, HDIM / BN);   // (391, 8)

  // fc0 -> LN(relu)
  k_gemm_bf16<<<gSmall, blk, 0, stream>>>(hB, fc0BT, fc0_b, attnF, nullptr,
                                          NNODES, CCH, CCH);
  k_layernorm<<<dim3(NNODES), blk, 0, stream>>>(attnF, ln_w, ln_b, hF, hB, 1);

  for (int layer = 0; layer < NLAYERS; ++layer) {
    k_zero_f32<<<dim3(1), blk, 0, stream>>>(ss, 2);
    k_zero_f32<<<dim3((HDIM + 255) / 256), blk, 0, stream>>>(ksum, HDIM);
    k_zero_f32<<<dim3((NH * DH * DH + 255) / 256), blk, 0, stream>>>(
        kvsFT, NH * DH * DH);

    const __bf16* Wqb = WqBT + (size_t)layer * HDIM * CCH;
    const __bf16* Wkb = WkBT + (size_t)layer * HDIM * CCH;
    const __bf16* Wvb = WvBT + (size_t)layer * HDIM * CCH;
    k_gemm_bf16<<<gProj, blk, 0, stream>>>(hB, Wqb, bq + layer * HDIM, nullptr,
                                           qB, NNODES, HDIM, CCH);
    k_gemm_bf16<<<gProj, blk, 0, stream>>>(hB, Wkb, bk + layer * HDIM, nullptr,
                                           kB, NNODES, HDIM, CCH);
    k_gemm_bf16<<<gProj, blk, 0, stream>>>(hB, Wvb, bv + layer * HDIM, nullptr,
                                           vB, NNODES, HDIM, CCH);

    k_sumsq<<<dim3(1024), blk, 0, stream>>>(qB, (long)NNODES * HDIM, ss + 0);
    k_sumsq<<<dim3(1024), blk, 0, stream>>>(kB, (long)NNODES * HDIM, ss + 1);
    k_colsum<<<dim3((NNODES + RPB - 1) / RPB), blk, 0, stream>>>(kB, NNODES,
                                                                 ksum);

    dim3 gKvs(4, NH, (NNODES + KCHUNK - 1) / KCHUNK);
    k_kvs<<<gKvs, blk, 0, stream>>>(kB, vB, kvsFT, NNODES);
    k_f32_to_bf16<<<dim3((NH * DH * DH + 255) / 256), blk, 0, stream>>>(
        kvsFT, kvsBT, NH * DH * DH);

    k_denom<<<dim3((NNODES * NH + 255) / 256), blk, 0, stream>>>(qB, ksum, ss,
                                                                 denom, NNODES);
    k_attn<<<gSmall, blk, 0, stream>>>(qB, vB, kvsBT, denom, ss, hF, attnF,
                                       NNODES);
    k_layernorm<<<dim3(NNODES), blk, 0, stream>>>(
        attnF, ln_w + (size_t)(layer + 1) * CCH,
        ln_b + (size_t)(layer + 1) * CCH, hF, hB, 0);
  }

  k_gemm_bf16<<<gSmall, blk, 0, stream>>>(hB, fcBT, fc_b, out, nullptr, NNODES,
                                          CCH, CCH);
}
